// Lstm_model_88012469829659
// MI455X (gfx1250) — matrix operand order repaired and checked by reading, not compiled, not run on hardware
//
#include <hip/hip_runtime.h>
#include <hip/hip_bf16.h>
#include <stdint.h>

// ---------------------------------------------------------------------------
// Model constants
// ---------------------------------------------------------------------------
#define NEGV      -10000.0f
#define NTAGS     27
#define TAG_START 25
#define TAG_STOP  26
#define HID       256
#define GATES     1024      // 4*HID
#define EMB_K     100
#define EMB_KP    128       // padded K for WMMA
#define SC_STRIDE 32        // padded tag count (scores / bptr stride)

// ---------------------------------------------------------------------------
// CDNA5 types
// ---------------------------------------------------------------------------
typedef __bf16        v16bf  __attribute__((ext_vector_type(16)));
typedef float         v8f    __attribute__((ext_vector_type(8)));
typedef unsigned int  u32x4  __attribute__((ext_vector_type(4)));
typedef int           i32x4  __attribute__((ext_vector_type(4)));
typedef int           i32x8  __attribute__((ext_vector_type(8)));

struct __align__(16) U128 { unsigned int x, y, z, w; };

#if defined(__gfx1250__) && __has_builtin(__builtin_amdgcn_tensor_load_to_lds)
#define USE_TDM 1
#else
#define USE_TDM 0
#endif

// ---------------------------------------------------------------------------
// bf16 helpers
// ---------------------------------------------------------------------------
__device__ __forceinline__ unsigned short f2bf(float f) {
    union { float f; unsigned int u; } x; x.f = f;
    unsigned int u = x.u;
    unsigned int r = (u + 0x7FFFu + ((u >> 16) & 1u)) >> 16;
    return (unsigned short)r;
}
__device__ __forceinline__ float fsigmoid(float x) {
    return 1.0f / (1.0f + __expf(-x));
}
__device__ __forceinline__ float ftanh(float x) {
    float e = __expf(2.0f * x);
    return (e - 1.0f) / (e + 1.0f);
}

// ---------------------------------------------------------------------------
// Kernel 1: convert + pad f32 weights -> bf16 [Np, Kp] (zeros in padding)
// ---------------------------------------------------------------------------
__global__ void cvt_pad_bf16(const float* __restrict__ src, unsigned short* __restrict__ dst,
                             int N, int K, int Kp, int Np) {
    int idx = blockIdx.x * blockDim.x + threadIdx.x;
    if (idx >= Np * Kp) return;
    int n = idx / Kp, k = idx % Kp;
    float v = (n < N && k < K) ? src[(size_t)n * K + k] : 0.0f;
    dst[idx] = f2bf(v);
}

// ---------------------------------------------------------------------------
// Kernel 2: embedding gather -> bf16 [T, EMB_KP], zero-padded K
// ---------------------------------------------------------------------------
__global__ void embed_gather(const int* __restrict__ sent, const float* __restrict__ emb,
                             unsigned short* __restrict__ x) {
    int t = blockIdx.x;
    int k = threadIdx.x;                     // blockDim.x == EMB_KP
    float v = (k < EMB_K) ? emb[(size_t)sent[t] * EMB_K + k] : 0.0f;
    x[(size_t)t * EMB_KP + k] = f2bf(v);
}

// ---------------------------------------------------------------------------
// Kernel 3: GEMM  C[M,N] = A[M,Kp](bf16) * W[N,Kp](bf16)^T + bias1 + bias2
// One wave per 16x16 tile, v_wmma_f32_16x16x32_bf16, fragments per ISA layout.
// ---------------------------------------------------------------------------
__global__ void gemm_bf16_wmma(const unsigned short* __restrict__ A,
                               const unsigned short* __restrict__ W,
                               const float* __restrict__ bias1,
                               const float* __restrict__ bias2,
                               float* __restrict__ C,
                               int M, int N, int Kp, int nbias) {
    const int lane = threadIdx.x & 31;
    const int wave = (blockIdx.x * blockDim.x + threadIdx.x) >> 5;
    const int tilesN = N >> 4;
    const int mt = wave / tilesN;
    const int nt = wave - mt * tilesN;
    if (mt >= (M >> 4)) return;

    const int half = lane >> 4;      // 0: lanes 0-15, 1: lanes 16-31
    const int l16  = lane & 15;

    const unsigned short* arow = A + (size_t)(mt * 16 + l16) * Kp;
    const unsigned short* wrow = W + (size_t)(nt * 16 + l16) * Kp;

    v8f acc = {};
    for (int k0 = 0; k0 < Kp; k0 += 32) {
        __builtin_prefetch(arow + k0 + 64, 0, 1);  // global_prefetch_b8
        union { v16bf v; U128 q[2]; } a, b;
        // A 16x32 fragment: e<8 -> K = k0 + 8*half + e ; e>=8 -> K = k0+16+8*half+(e-8)
        a.q[0] = *(const U128*)(arow + k0 + half * 8);
        a.q[1] = *(const U128*)(arow + k0 + 16 + half * 8);
        // B 32x16 fragment (W^T): e<8 -> K = k0 + 8*half + e ; e>=8 -> K = k0+16+8*half+(e-8), N = l16
        b.q[0] = *(const U128*)(wrow + k0 + half * 8);
        b.q[1] = *(const U128*)(wrow + k0 + 16 + half * 8);
        acc = __builtin_amdgcn_wmma_f32_16x16x32_bf16(
                  false, a.v, false, b.v, (short)0, acc, false, false);
    }

    const int n = nt * 16 + l16;
    float bv = 0.0f;
    if (bias1 && n < nbias) bv += bias1[n];
    if (bias2 && n < nbias) bv += bias2[n];

    // D layout: VGPR r -> row mt*16 + r + 8*half, col n
    float* cptr = C + (size_t)(mt * 16 + half * 8) * N + n;
#pragma unroll
    for (int r = 0; r < 8; ++r)
        cptr[(size_t)r * N] = acc[r] + bv;
}

// ---------------------------------------------------------------------------
// Kernel 4: persistent WMMA LSTM recurrence.
// One workgroup (1024 thr = 32 waves on one WGP). W_hh^T (512 KB bf16) lives
// permanently in the register file as WMMA B-fragments: each wave owns two
// 16-column gate tiles x 8 K-fragments = 128 VGPRs/lane. Per step each wave
// rebuilds 8 A-fragments from h in LDS (two broadcast ds_load_b128 each; all
// 16 M-rows replicate the same 32-wide h slice, so D row 0 is the matvec) and
// issues 16 v_wmma_f32_16x16x32_bf16. The 4KB xp[t] row is double-buffered
// into LDS by the Tensor Data Mover, synchronized with s_wait_tensorcnt.
// ---------------------------------------------------------------------------
__global__ __launch_bounds__(1024) void lstm_rec_wmma(const float* __restrict__ xp,
                                                      const unsigned short* __restrict__ whh_bf,
                                                      unsigned short* __restrict__ hout_bf,
                                                      int T) {
    __shared__ __align__(16) unsigned short hsb[HID];   // h as bf16
    __shared__ float gates[GATES];
    __shared__ float xbuf[2][GATES];

    const int tid  = threadIdx.x;
    const int lane = tid & 31;
    const int w    = tid >> 5;       // wave id 0..31
    const int half = lane >> 4;
    const int l16  = lane & 15;

    // ---- load this wave's share of W_hh^T as 16 B-fragments (128 VGPRs) ----
    // B fragment (K=32 x N=16): e<8 -> K = kf*32 + 8*half + e ; e>=8 -> K = kf*32 + 16 + 8*half + (e-8), N = l16.
    v16bf bfrag[16];
#pragma unroll
    for (int i = 0; i < 2; ++i) {
        const unsigned short* wrow =
            whh_bf + (size_t)((2 * w + i) * 16 + l16) * HID;
#pragma unroll
        for (int kf = 0; kf < 8; ++kf) {
            union { v16bf v; U128 q[2]; } b;
            b.q[0] = *(const U128*)(wrow + kf * 32 + half * 8);
            b.q[1] = *(const U128*)(wrow + kf * 32 + 16 + half * 8);
            bfrag[i * 8 + kf] = b.v;
        }
    }

    if (tid < HID) hsb[tid] = 0;     // h = 0
    float c = 0.0f;

#if USE_TDM
    const unsigned int lds_x0 = (unsigned int)(uintptr_t)&xbuf[0][0];
    const unsigned int lds_x1 = (unsigned int)(uintptr_t)&xbuf[1][0];
    // 1D D# : 4KB tile (512 x 8B), tensor_dim0 = stride = 512, type=2.
    auto tdm_row = [&](const float* gp, unsigned int lds_off) {
        unsigned long long ga = (unsigned long long)(uintptr_t)gp;
        u32x4 g0;
        g0[0] = 1u;                                   // count=1, user descriptor
        g0[1] = lds_off;                              // lds_addr
        g0[2] = (unsigned int)(ga & 0xFFFFFFFFu);     // global_addr[31:0]
        g0[3] = (unsigned int)((ga >> 32) & 0x1FFFFFFu) | (2u << 30); // addr[56:32]|type=2
        i32x8 g1;
        g1[0] = (3 << 16);        // data_size = 8B, no multicast
        g1[1] = (512 << 16);      // tensor_dim0[15:0] = 512
        g1[2] = (1 << 16);        // tensor_dim1 = 1
        g1[3] = (512 << 16);      // tile_dim0 = 512
        g1[4] = 0;                // tile_dim1/2 unused
        g1[5] = 512;              // tensor_dim0_stride
        g1[6] = 0; g1[7] = 0;
        i32x4 z = {};
#if defined(__clang_major__) && (__clang_major__ >= 23)
        i32x8 z8 = {};
        __builtin_amdgcn_tensor_load_to_lds(g0, g1, z, z, z8, 0);
#else
        __builtin_amdgcn_tensor_load_to_lds(g0, g1, z, z, 0);
#endif
    };
    if (tid < 32) {               // single wave drives the TDM
        tdm_row(xp, lds_x0);
        __builtin_amdgcn_s_wait_tensorcnt(0);
    }
#else
    xbuf[0][tid] = xp[tid];
#endif
    __syncthreads();

    for (int t = 0; t < T; ++t) {
        const int cur = t & 1;
#if USE_TDM
        if (tid < 32 && (t + 1) < T)
            tdm_row(xp + (size_t)(t + 1) * GATES, cur ? lds_x0 : lds_x1);
#else
        __builtin_prefetch(xp + (size_t)(t + 1) * GATES + tid, 0, 0);
#endif
        // ---- h . W_hh^T via WMMA: A rows all replicate h[kf*32 .. kf*32+31] ----
        v8f acc0 = {}, acc1 = {};
#pragma unroll
        for (int kf = 0; kf < 8; ++kf) {
            union { v16bf v; U128 q[2]; } a;
            // e<8 -> K = kf*32 + 8*half + e ; e>=8 -> K = kf*32 + 16 + 8*half + (e-8)
            a.q[0] = *(const U128*)&hsb[kf * 32 + half * 8];
            a.q[1] = *(const U128*)&hsb[kf * 32 + 16 + half * 8];
            acc0 = __builtin_amdgcn_wmma_f32_16x16x32_bf16(
                       false, a.v, false, bfrag[kf], (short)0, acc0, false, false);
            acc1 = __builtin_amdgcn_wmma_f32_16x16x32_bf16(
                       false, a.v, false, bfrag[8 + kf], (short)0, acc1, false, false);
        }
        // D row 0 (M=0) lives in VGPR0 of lanes 0..15 -> gate pre-activations.
        if (lane < 16) {
            const int n0 = 32 * w + lane;
            const int n1 = n0 + 16;
#if USE_TDM
            gates[n0] = acc0[0] + xbuf[cur][n0];
            gates[n1] = acc1[0] + xbuf[cur][n1];
#else
            gates[n0] = acc0[0] + xp[(size_t)t * GATES + n0];
            gates[n1] = acc1[0] + xp[(size_t)t * GATES + n1];
#endif
        }
        __syncthreads();

        if (tid < HID) {
            float ig = fsigmoid(gates[tid]);
            float fg = fsigmoid(gates[HID + tid]);
            float gg = ftanh(gates[2 * HID + tid]);
            float og = fsigmoid(gates[3 * HID + tid]);
            c = fg * c + ig * gg;
            float h = og * ftanh(c);
            hsb[tid] = f2bf(h);
            hout_bf[(size_t)t * HID + tid] = f2bf(h);
        }
#if USE_TDM
        if (tid < 32) __builtin_amdgcn_s_wait_tensorcnt(0);  // row t+1 landed
#endif
        __syncthreads();
    }
}

// ---------------------------------------------------------------------------
// Kernel 5: Viterbi decode. One wave; v in LDS, backpointers u8 in workspace.
// ---------------------------------------------------------------------------
__global__ void viterbi(const float* __restrict__ scores,   // [T, SC_STRIDE]
                        const int* __restrict__ mask,       // [T]
                        const float* __restrict__ trans,    // [27,27]
                        unsigned char* __restrict__ bptr,   // [T, SC_STRIDE]
                        float* __restrict__ out, int T) {
    const int j = threadIdx.x;
    __shared__ float v[32];
    __shared__ float tr[NTAGS * NTAGS];
    __shared__ float term[32];

    for (int i = j; i < NTAGS * NTAGS; i += 32) tr[i] = trans[i];
    v[j] = (j == TAG_START) ? 0.0f : NEGV;
    __syncthreads();

    for (int t = 0; t < T; ++t) {
        float best = -3.4e38f; int bi = 0;
        if (j < NTAGS) {
#pragma unroll 1
            for (int i = 0; i < NTAGS; ++i) {
                float s = v[i] + tr[i * NTAGS + j];
                if (s > best) { best = s; bi = i; }
            }
        }
        const int m = mask[t];
        const float vn = best + scores[(size_t)t * SC_STRIDE + j];
        __syncthreads();
        if (j < NTAGS) {
            v[j] = m ? vn : v[j];
            bptr[(size_t)t * SC_STRIDE + j] = (unsigned char)(m ? bi : j);
        }
        __syncthreads();
    }

    term[j] = (j < NTAGS) ? (v[j] + tr[j * NTAGS + TAG_STOP]) : -3.4e38f;
    __syncthreads();
    __threadfence_block();
    if (j == 0) {
        int best = 0; float bs = term[0];
        for (int i = 1; i < NTAGS; ++i)
            if (term[i] > bs) { bs = term[i]; best = i; }
        out[0] = bs;
        int tag = best;
        for (int t = T - 1; t >= 0; --t) {
            if (t >= 64) __builtin_prefetch(bptr + (size_t)(t - 64) * SC_STRIDE, 0, 0);
            out[1 + t] = (float)tag;
            tag = bptr[(size_t)t * SC_STRIDE + tag];
        }
    }
}

// ---------------------------------------------------------------------------
// Host launcher
// ---------------------------------------------------------------------------
extern "C" void kernel_launch(void* const* d_in, const int* in_sizes, int n_in,
                              void* d_out, int out_size, void* d_ws, size_t ws_size,
                              hipStream_t stream) {
    const int*   sent  = (const int*)  d_in[0];
    const int*   amask = (const int*)  d_in[1];
    const float* emb   = (const float*)d_in[2];
    const float* w_ih0 = (const float*)d_in[3];
    const float* w_hh0 = (const float*)d_in[4];
    const float* b_ih0 = (const float*)d_in[5];
    const float* b_hh0 = (const float*)d_in[6];
    const float* w_ih1 = (const float*)d_in[7];
    const float* w_hh1 = (const float*)d_in[8];
    const float* b_ih1 = (const float*)d_in[9];
    const float* b_hh1 = (const float*)d_in[10];
    const float* w_lin = (const float*)d_in[11];
    const float* b_lin = (const float*)d_in[12];
    const float* trans = (const float*)d_in[13];
    float* out = (float*)d_out;
    const int T = in_sizes[0];

    // ---- workspace layout ----
    char* ws = (char*)d_ws;
    size_t off = 0;
    auto take = [&](size_t bytes) -> char* {
        char* p = ws + off;
        off = (off + bytes + 255) & ~(size_t)255;
        return p;
    };
    unsigned short* x_bf    = (unsigned short*)take((size_t)T * EMB_KP * 2);
    unsigned short* wih0_bf = (unsigned short*)take((size_t)GATES * EMB_KP * 2);
    unsigned short* whh0_bf = (unsigned short*)take((size_t)GATES * HID * 2);
    unsigned short* wih1_bf = (unsigned short*)take((size_t)GATES * HID * 2);
    unsigned short* whh1_bf = (unsigned short*)take((size_t)GATES * HID * 2);
    unsigned short* wlin_bf = (unsigned short*)take((size_t)SC_STRIDE * HID * 2);
    float*          xpbuf   = (float*)take((size_t)T * GATES * 4);
    unsigned short* h0_bf   = (unsigned short*)take((size_t)T * HID * 2);
    unsigned short* h1_bf   = (unsigned short*)take((size_t)T * HID * 2);
    float*          scores  = (float*)take((size_t)T * SC_STRIDE * 4);
    unsigned char*  bptr    = (unsigned char*)take((size_t)T * SC_STRIDE);

    // ---- weight conversion to padded bf16 ----
    auto cvt = [&](const float* src, unsigned short* dst, int N, int K, int Kp, int Np) {
        int tot = Np * Kp;
        cvt_pad_bf16<<<(tot + 255) / 256, 256, 0, stream>>>(src, dst, N, K, Kp, Np);
    };
    cvt(w_ih0, wih0_bf, GATES, EMB_K, EMB_KP, GATES);
    cvt(w_hh0, whh0_bf, GATES, HID,   HID,    GATES);
    cvt(w_ih1, wih1_bf, GATES, HID,   HID,    GATES);
    cvt(w_hh1, whh1_bf, GATES, HID,   HID,    GATES);
    cvt(w_lin, wlin_bf, NTAGS, HID,   HID,    SC_STRIDE);

    // ---- embedding gather ----
    embed_gather<<<T, EMB_KP, 0, stream>>>(sent, emb, x_bf);

    auto gemm = [&](const unsigned short* A, const unsigned short* W,
                    const float* bias1, const float* bias2,
                    float* C, int M, int N, int Kp, int nbias) {
        int waves = (M / 16) * (N / 16);
        int blocks = (waves + 7) / 8;          // 8 wave32s per 256-thread block
        gemm_bf16_wmma<<<blocks, 256, 0, stream>>>(A, W, bias1, bias2, C, M, N, Kp, nbias);
    };

    // ---- layer 0: xp = x @ w_ih0^T + b_ih0 + b_hh0 ; recurrence ----
    gemm(x_bf, wih0_bf, b_ih0, b_hh0, xpbuf, T, GATES, EMB_KP, GATES);
    lstm_rec_wmma<<<1, 1024, 0, stream>>>(xpbuf, whh0_bf, h0_bf, T);

    // ---- layer 1 ----
    gemm(h0_bf, wih1_bf, b_ih1, b_hh1, xpbuf, T, GATES, HID, GATES);
    lstm_rec_wmma<<<1, 1024, 0, stream>>>(xpbuf, whh1_bf, h1_bf, T);

    // ---- emission scores: [T, 32], cols >= 27 are zero-weight ----
    gemm(h1_bf, wlin_bf, b_lin, nullptr, scores, T, SC_STRIDE, HID, NTAGS);

    // ---- Viterbi decode ----
    viterbi<<<1, 32, 0, stream>>>(scores, amask, trans, bptr, out, T);
}
